// CachedMultiHeadAttention_74706661147057
// MI455X (gfx1250) — compile-verified, matrix operand order checked by reading, not run on hardware
//
#include <hip/hip_runtime.h>

// ---------------------------------------------------------------------------
// Types for CDNA5 WMMA (wave32). 16x16x32 bf16 -> f32.
// ---------------------------------------------------------------------------
typedef __attribute__((ext_vector_type(16))) __bf16 v16bf;
typedef __attribute__((ext_vector_type(8)))  __bf16 v8bf;
typedef __attribute__((ext_vector_type(4)))  __bf16 v4bf;
typedef __attribute__((ext_vector_type(8)))  float  v8f;
typedef __attribute__((ext_vector_type(4)))  unsigned int v4u;
typedef __attribute__((ext_vector_type(8)))  int v8i;
typedef __attribute__((ext_vector_type(4)))  int v4i;

#define NH 16
#define HD 64
#define SEQ 2048
#define BATCH 2
#define HID 1024
#define H3 3072

#define HAS_TDM __has_builtin(__builtin_amdgcn_tensor_load_to_lds)

__device__ __forceinline__ v4bf cvt4(float4 v) {
    v4bf r;
    r[0] = (__bf16)v.x; r[1] = (__bf16)v.y; r[2] = (__bf16)v.z; r[3] = (__bf16)v.w;
    return r;
}

__device__ __forceinline__ v8bf ld8(const __bf16* p) {
    return *reinterpret_cast<const v8bf*>(p);
}

__device__ __forceinline__ v16bf mk16(v8bf lo, v8bf hi) {
    v16bf r;
#pragma unroll
    for (int i = 0; i < 8; ++i) { r[i] = lo[i]; r[i + 8] = hi[i]; }
    return r;
}

__device__ __forceinline__ v8f wmma_bf16(v16bf a, v16bf b, v8f c) {
    return __builtin_amdgcn_wmma_f32_16x16x32_bf16(
        /*neg_a=*/false, a, /*neg_b=*/false, b,
        /*c_mod=*/(short)0, c, /*reuse_a=*/false, /*reuse_b=*/false);
}

__device__ __forceinline__ v8f vzero8() {
    v8f z;
#pragma unroll
    for (int i = 0; i < 8; ++i) z[i] = 0.0f;
    return z;
}

#if HAS_TDM
// ---------------------------------------------------------------------------
// TDM: load a 64x64 bf16 tile (row stride `srcStride` elements) into LDS with
// a post-pad of 4 DWORDs after every 32 DWORDs (=> LDS row stride 72 bf16).
// D# layout per CDNA5 ISA sec 8.3/8.4. One wave issues; TENSORcnt tracks.
// This toolchain exposes the 6-arg builtin: (v4u, v8i, v4i, v4i, v8i, cpol).
// ---------------------------------------------------------------------------
__device__ __forceinline__ void tdm_load_64x64_bf16(const __bf16* src, __bf16* dstLds,
                                                    unsigned srcStride) {
    const unsigned long long ga = (unsigned long long)(const void*)src;
    const unsigned lds = (unsigned)(unsigned long long)(void*)dstLds;
    v4u g0;
    g0[0] = 1u;                                        // count=1 valid descriptor
    g0[1] = lds;                                       // lds_addr
    g0[2] = (unsigned)(ga & 0xffffffffu);              // global_addr[31:0]
    g0[3] = (unsigned)((ga >> 32) & 0x01ffffffu)       // global_addr[56:32]
            | (2u << 30);                              // type = 2 (image)
    v8i g1;
    g1[0] = (1 << 16)      // data_size = 1 -> 2 bytes
          | (1 << 20)      // pad_enable
          | (4 << 22)      // pad_interval: 4 -> every 32 DWORDs (64 bf16)
          | (3 << 25);     // pad_amount: 3 -> 4 DWORDs (8 bf16)  => stride 72
    g1[1] = (64u << 16);                   // tensor_dim0[15:0] in [63:48]
    g1[2] = (64u << 16);                   // dim0 hi=0 | tensor_dim1 lo = 64
    g1[3] = (64u << 16);                   // dim1 hi=0 | tile_dim0 = 64
    g1[4] = 64;                            // tile_dim1 = 64 | tile_dim2 = 0
    g1[5] = (int)srcStride;                // tensor_dim0_stride[31:0]
    g1[6] = 0;                             // stride hi | dim1_stride lo
    g1[7] = 0;
    v4i g2 = {0, 0, 0, 0};
    v4i g3 = {0, 0, 0, 0};
    v8i g4 = {0, 0, 0, 0, 0, 0, 0, 0};
    __builtin_amdgcn_tensor_load_to_lds(g0, g1, g2, g3, g4, 0);
}
#endif

// ---------------------------------------------------------------------------
// GEMM: C[M,N] = A[M,K] * B[N,K]^T + bias[N]   (both contract along last dim)
// Block tile 128x128, 256 threads = 8 waves (4 row-waves x 2 col-waves).
// K-stage = 64: 16 WMMAs per wave between barriers.
// A_BF16: A operand already bf16 (no convert). C_BF16: store output as bf16.
// ---------------------------------------------------------------------------
#define GSTRIDE 72   // 64 bf16 + 8 pad (144 bytes/row, 16B aligned)

template <int A_BF16, int C_BF16>
__global__ __launch_bounds__(256)
void gemm_wmma(const void* __restrict__ Aptr, const float* __restrict__ Bw,
               const float* __restrict__ bias, void* __restrict__ Cptr,
               int M, int N, int K) {
    __shared__ alignas(16) __bf16 sA[128 * GSTRIDE];
    __shared__ alignas(16) __bf16 sB[128 * GSTRIDE];

    const int tid   = threadIdx.x;
    const int wave  = tid >> 5;
    const int lane  = tid & 31;
    const int half  = lane >> 4;
    const int ln16  = lane & 15;
    const int waveM = wave & 3;     // 0..3  -> 32-row slab
    const int waveN = wave >> 2;    // 0..1  -> 64-col slab
    const int blockM = blockIdx.y * 128;
    const int blockN = blockIdx.x * 128;

    const int lr = tid >> 4;         // 0..15 loader row
    const int lc = (tid & 15) * 4;   // 0..60 loader col (elements)

    v8f acc[2][4];
#pragma unroll
    for (int i = 0; i < 2; ++i)
#pragma unroll
        for (int j = 0; j < 4; ++j) acc[i][j] = vzero8();

    for (int k0 = 0; k0 < K; k0 += 64) {
        // --- cooperative stage load (128 x 64 per tile) ---
#pragma unroll
        for (int t = 0; t < 8; ++t) {
            const int row = lr + 16 * t;
            if (A_BF16) {
                const __bf16* A = (const __bf16*)Aptr;
                *reinterpret_cast<v4bf*>(&sA[row * GSTRIDE + lc]) =
                    *reinterpret_cast<const v4bf*>(A + (size_t)(blockM + row) * K + k0 + lc);
            } else {
                const float* A = (const float*)Aptr;
                float4 va = *reinterpret_cast<const float4*>(
                    A + (size_t)(blockM + row) * K + k0 + lc);
                *reinterpret_cast<v4bf*>(&sA[row * GSTRIDE + lc]) = cvt4(va);
            }
            float4 vb = *reinterpret_cast<const float4*>(
                Bw + (size_t)(blockN + row) * K + k0 + lc);
            *reinterpret_cast<v4bf*>(&sB[row * GSTRIDE + lc]) = cvt4(vb);
        }
        // prefetch next stage while this one computes
        if (k0 + 64 < K) {
            __builtin_prefetch(Bw + (size_t)(blockN + lr) * K + k0 + 64 + lc, 0, 1);
        }
        __syncthreads();

        // --- 2 k-substeps x (2x4) accumulators = 16 WMMAs per wave ---
#pragma unroll
        for (int kk = 0; kk < 2; ++kk) {
            v16bf af[2];
#pragma unroll
            for (int i = 0; i < 2; ++i) {
                const __bf16* pa = &sA[(waveM * 32 + i * 16 + ln16) * GSTRIDE + kk * 32];
                af[i] = mk16(ld8(pa + half * 8), ld8(pa + 16 + half * 8));
            }
            v16bf bfm[4];
#pragma unroll
            for (int j = 0; j < 4; ++j) {
                const __bf16* pb = &sB[(waveN * 64 + j * 16 + ln16) * GSTRIDE + kk * 32];
                bfm[j] = mk16(ld8(pb + half * 8), ld8(pb + 16 + half * 8));
            }
#pragma unroll
            for (int i = 0; i < 2; ++i)
#pragma unroll
                for (int j = 0; j < 4; ++j)
                    acc[i][j] = wmma_bf16(af[i], bfm[j], acc[i][j]);
        }
        __syncthreads();
    }

    // --- epilogue: bias + store (C fragment: row = base + r + 8*half, col = lane%16)
#pragma unroll
    for (int i = 0; i < 2; ++i) {
#pragma unroll
        for (int j = 0; j < 4; ++j) {
            const int row0 = blockM + waveM * 32 + i * 16 + half * 8;
            const int col  = blockN + waveN * 64 + j * 16 + ln16;
            const float bv = bias[col];
#pragma unroll
            for (int r = 0; r < 8; ++r) {
                const float v = acc[i][j][r] + bv;
                if (C_BF16)
                    ((__bf16*)Cptr)[(size_t)(row0 + r) * N + col] = (__bf16)v;
                else
                    ((float*)Cptr)[(size_t)(row0 + r) * N + col] = v;
            }
        }
    }
}

// ---------------------------------------------------------------------------
// Flash attention on bf16 qkv: block = (q-tile 128, head, batch).
// K tiles via TDM (tensor_load_to_lds) when available; V transposed-scatter.
// ---------------------------------------------------------------------------
#define KSTRIDE 72  // 64 bf16 + 8 pad (144 bytes, 16B aligned)

__global__ __launch_bounds__(256)
void attn_kernel(const __bf16* __restrict__ qkv, const float* __restrict__ mask,
                 __bf16* __restrict__ ctx) {
    __shared__ alignas(16) __bf16 sK [64 * KSTRIDE];
    __shared__ alignas(16) __bf16 sVt[64 * KSTRIDE];
    __shared__ alignas(16) __bf16 sP [8 * 16 * KSTRIDE];

    const int tid  = threadIdx.x;
    const int wave = tid >> 5;
    const int lane = tid & 31;
    const int half = lane >> 4;
    const int ln16 = lane & 15;

    const int q0 = blockIdx.x * 128;
    const int h  = blockIdx.y;
    const int b  = blockIdx.z;
    const float scale = 0.125f;  // HD^-0.5

    // --- Q fragments: direct bf16 b128 global loads (A-layout chunks) ---
    const __bf16* qp = qkv + (size_t)(b * SEQ + q0 + wave * 16 + ln16) * H3 + h * HD;
    v16bf qf[2];
#pragma unroll
    for (int kk = 0; kk < 2; ++kk)
        qf[kk] = mk16(ld8(qp + kk * 32 + half * 8), ld8(qp + kk * 32 + 16 + half * 8));

    float mrow[8], lrow[8];
#pragma unroll
    for (int r = 0; r < 8; ++r) { mrow[r] = -1.0e30f; lrow[r] = 0.0f; }
    v8f oacc[4];
#pragma unroll
    for (int j = 0; j < 4; ++j) oacc[j] = vzero8();

    const int lr  = tid >> 4;        // 0..15 loader row
    const int lc4 = (tid & 15) * 4;  // 0..60 loader col

    for (int kv0 = 0; kv0 < SEQ; kv0 += 64) {
        __syncthreads();  // previous tile fully consumed

        // --- K tile: Tensor Data Mover (one wave issues the whole 64x64 DMA) ---
#if HAS_TDM
        if (wave == 0) {
            tdm_load_64x64_bf16(qkv + (size_t)(b * SEQ + kv0) * H3 + HID + h * HD,
                                sK, H3);
        }
#else
#pragma unroll
        for (int t = 0; t < 4; ++t) {
            const int row = lr + 16 * t;
            const size_t tok = (size_t)(b * SEQ + kv0 + row) * H3 + h * HD;
            *reinterpret_cast<v4bf*>(&sK[row * KSTRIDE + lc4]) =
                *reinterpret_cast<const v4bf*>(qkv + tok + HID + lc4);
        }
#endif
        // --- V tile: transposed scatter (bf16 -> Vt[d][n]) ---
#pragma unroll
        for (int t = 0; t < 4; ++t) {
            const int row = lr + 16 * t;
            const size_t tok = (size_t)(b * SEQ + kv0 + row) * H3 + h * HD;
            v4bf vv = *reinterpret_cast<const v4bf*>(qkv + tok + 2 * HID + lc4);
            sVt[(lc4 + 0) * KSTRIDE + row] = vv[0];
            sVt[(lc4 + 1) * KSTRIDE + row] = vv[1];
            sVt[(lc4 + 2) * KSTRIDE + row] = vv[2];
            sVt[(lc4 + 3) * KSTRIDE + row] = vv[3];
        }
#if HAS_TDM
        if (wave == 0) __builtin_amdgcn_s_wait_tensorcnt(0);
#endif
        __syncthreads();

        // --- scores: 16 q-rows x 64 keys = 4 fragments, K-dim = 64 (2 steps) ---
        v8f sfr[4];
#pragma unroll
        for (int j = 0; j < 4; ++j) {
            const __bf16* kb = &sK[(j * 16 + ln16) * KSTRIDE];
            v16bf b0 = mk16(ld8(kb + half * 8),      ld8(kb + 16 + half * 8));
            v16bf b1 = mk16(ld8(kb + 32 + half * 8), ld8(kb + 48 + half * 8));
            v8f z = vzero8();
            z = wmma_bf16(qf[0], b0, z);
            z = wmma_bf16(qf[1], b1, z);
            sfr[j] = z;
        }

        float mcol[4];
#pragma unroll
        for (int j = 0; j < 4; ++j)
            mcol[j] = mask[b * SEQ + kv0 + j * 16 + ln16];

#pragma unroll
        for (int j = 0; j < 4; ++j)
#pragma unroll
            for (int r = 0; r < 8; ++r)
                sfr[j][r] = sfr[j][r] * scale + mcol[j];

        // --- online softmax per row (reduce across 16-lane half-groups) ---
#pragma unroll
        for (int r = 0; r < 8; ++r) {
            float v = -1.0e30f;
#pragma unroll
            for (int j = 0; j < 4; ++j) v = fmaxf(v, sfr[j][r]);
            v = fmaxf(v, __shfl_xor(v, 1, 32));
            v = fmaxf(v, __shfl_xor(v, 2, 32));
            v = fmaxf(v, __shfl_xor(v, 4, 32));
            v = fmaxf(v, __shfl_xor(v, 8, 32));
            const float mnew  = fmaxf(mrow[r], v);
            const float alpha = __expf(mrow[r] - mnew);
            mrow[r] = mnew;
            float sum = 0.0f;
#pragma unroll
            for (int j = 0; j < 4; ++j) {
                float p = __expf(sfr[j][r] - mnew);
                sfr[j][r] = p;
                sum += p;
            }
            sum += __shfl_xor(sum, 1, 32);
            sum += __shfl_xor(sum, 2, 32);
            sum += __shfl_xor(sum, 4, 32);
            sum += __shfl_xor(sum, 8, 32);
            lrow[r] = lrow[r] * alpha + sum;
#pragma unroll
            for (int j = 0; j < 4; ++j) oacc[j][r] *= alpha;
        }

        // --- C-layout -> A-layout via per-wave LDS scratch (same-wave: in-order)
        __bf16* pw = &sP[wave * 16 * KSTRIDE];
#pragma unroll
        for (int r = 0; r < 8; ++r)
#pragma unroll
            for (int j = 0; j < 4; ++j)
                pw[(r + 8 * half) * KSTRIDE + j * 16 + ln16] = (__bf16)sfr[j][r];

        v16bf pf[2];
#pragma unroll
        for (int f = 0; f < 2; ++f) {
            const __bf16* pp = &sP[wave * 16 * KSTRIDE + ln16 * KSTRIDE + f * 32];
            pf[f] = mk16(ld8(pp + half * 8), ld8(pp + 16 + half * 8));
        }

        // --- O += P * V  (contract over keys; Vt[d][n] rows contiguous in n) ---
#pragma unroll
        for (int j = 0; j < 4; ++j) {
            const __bf16* vb = &sVt[(j * 16 + ln16) * KSTRIDE];
            v16bf vB0 = mk16(ld8(vb + half * 8),      ld8(vb + 16 + half * 8));
            v16bf vB1 = mk16(ld8(vb + 32 + half * 8), ld8(vb + 48 + half * 8));
            oacc[j] = wmma_bf16(pf[0], vB0, oacc[j]);
            oacc[j] = wmma_bf16(pf[1], vB1, oacc[j]);
        }
    }

    // --- finalize: O / l, write bf16 ctx[b, s, h*64+d] (A-side of out-proj) ---
#pragma unroll
    for (int r = 0; r < 8; ++r) {
        const float inv = 1.0f / lrow[r];
        const int row = q0 + wave * 16 + r + 8 * half;
#pragma unroll
        for (int j = 0; j < 4; ++j) {
            const int col = h * HD + j * 16 + ln16;
            ctx[(size_t)(b * SEQ + row) * HID + col] = (__bf16)(oacc[j][r] * inv);
        }
    }
}

// ---------------------------------------------------------------------------
// Launch
// ---------------------------------------------------------------------------
extern "C" void kernel_launch(void* const* d_in, const int* in_sizes, int n_in,
                              void* d_out, int out_size, void* d_ws, size_t ws_size,
                              hipStream_t stream) {
    (void)in_sizes; (void)n_in; (void)out_size; (void)ws_size;
    const float* hidden = (const float*)d_in[0];
    const float* mask   = (const float*)d_in[1];
    const float* w_qkv  = (const float*)d_in[2];
    const float* b_qkv  = (const float*)d_in[3];
    const float* w_out  = (const float*)d_in[4];
    const float* b_out  = (const float*)d_in[5];
    float* out = (float*)d_out;

    __bf16* qkvb = (__bf16*)d_ws;                                               // [4096, 3072] bf16
    __bf16* ctxb = (__bf16*)((char*)d_ws + (size_t)4096 * H3 * sizeof(__bf16)); // [4096, 1024] bf16

    dim3 blk(256);
    // QKV projection: fp32 A x fp32 B -> bf16 C
    gemm_wmma<0, 1><<<dim3(H3 / 128, (BATCH * SEQ) / 128), blk, 0, stream>>>(
        hidden, w_qkv, b_qkv, qkvb, BATCH * SEQ, H3, HID);
    // Attention: 16 q-tiles x 16 heads x 2 batch (bf16 in, bf16 out)
    attn_kernel<<<dim3(SEQ / 128, NH, BATCH), blk, 0, stream>>>(qkvb, mask, ctxb);
    // Output projection: bf16 A x fp32 B -> fp32 C
    gemm_wmma<1, 0><<<dim3(HID / 128, (BATCH * SEQ) / 128), blk, 0, stream>>>(
        ctxb, w_out, b_out, out, BATCH * SEQ, HID, HID);
}
